// EncBlock_36258113913401
// MI455X (gfx1250) — matrix operand order repaired and checked by reading, not compiled, not run on hardware
//
#include <hip/hip_runtime.h>
#include <cstdint>

#define D_MODEL 1024
#define N_HEADS 16
#define HEAD_DK 64
#define HID_DIM 4096
#define SEQ_LEN 2048
#define EPS_LN 1e-5f

typedef __attribute__((ext_vector_type(16))) __bf16 v16bf;
typedef __attribute__((ext_vector_type(8)))  float  v8f;

union FragU { v16bf v; uint32_t u[8]; };

__device__ __forceinline__ v8f wmma_bf16f32(v16bf a, v16bf b, v8f c) {
  // D = A(16x32 bf16) * B(32x16 bf16) + C(16x16 f32)
  return __builtin_amdgcn_wmma_f32_16x16x32_bf16(false, a, false, b, (short)0, c, false, false);
}

// A-matrix (16-bit, 16x32) layout: lane half picks K band; VGPR i holds a K pair.
// Returns dword index into a K-contiguous bf16 row.
__device__ __forceinline__ int a_k2(int i, int hf) { return i + hf * 4 + (i >= 4 ? 4 : 0); }

// ---- CDNA5 async global->LDS copy (ASYNCcnt path, CDNA5 ISA)
__device__ __forceinline__ void async_ld_b128(uint32_t lds_off, const void* gptr) {
  asm volatile("global_load_async_to_lds_b128 %0, %1, off"
               :: "v"(lds_off), "v"((uint64_t)(uintptr_t)gptr)
               : "memory");
}
__device__ __forceinline__ void wait_async_le4() {
  asm volatile("s_wait_asynccnt 0x4" ::: "memory");
}
__device__ __forceinline__ void wait_async_0() {
  asm volatile("s_wait_asynccnt 0x0" ::: "memory");
}

// ---------------------------------------------------------------- f32 -> bf16
__global__ __launch_bounds__(256) void f2bf_kernel(const float* __restrict__ src,
                                                   __bf16* __restrict__ dst, int n) {
  int i = (blockIdx.x * 256 + threadIdx.x) * 4;
  if (i < n) {
    float4 f = *(const float4*)(src + i);
    dst[i + 0] = (__bf16)f.x;
    dst[i + 1] = (__bf16)f.y;
    dst[i + 2] = (__bf16)f.z;
    dst[i + 3] = (__bf16)f.w;
  }
}

// ------------------------------------------- f32 [K,N] -> bf16 transposed [N,K]
__global__ __launch_bounds__(256) void f2bf_t_kernel(const float* __restrict__ src,
                                                     __bf16* __restrict__ dst,
                                                     int K, int N) {
  __shared__ float t[32][33];
  const int k0 = blockIdx.y * 32, n0 = blockIdx.x * 32;
  const int r = threadIdx.x >> 3, c4 = (threadIdx.x & 7) * 4;
  float4 f = *(const float4*)&src[(size_t)(k0 + r) * N + n0 + c4];
  t[r][c4 + 0] = f.x; t[r][c4 + 1] = f.y; t[r][c4 + 2] = f.z; t[r][c4 + 3] = f.w;
  __syncthreads();
#pragma unroll
  for (int j = 0; j < 4; ++j)
    dst[(size_t)(n0 + r) * K + k0 + c4 + j] = (__bf16)t[c4 + j][r];
}

// ---------------------------------------------------------------- WMMA GEMM
// C[M,N] = act(A[M,K] * Wt[N,K]^T + bias). A,Wt bf16, K-contiguous rows.
// WG tile 128x128, 8 waves (4x2), wave tile 32x64 = 2x4 WMMA tiles.
// Double-buffered LDS fed by async global->LDS b128 copies.
__global__ __launch_bounds__(256) void gemm_bf16_kernel(
    const __bf16* __restrict__ A, const __bf16* __restrict__ Wt,
    const float* __restrict__ bias, float* __restrict__ outF,
    __bf16* __restrict__ outB, int M, int N, int K, int relu) {
  __shared__ __bf16 Alds[2][128 * 32];   // [row][k]
  __shared__ __bf16 Blds[2][128 * 32];   // [col][k]  (Wt rows)

  const int tid = threadIdx.x;
  const int lane = tid & 31, wv = tid >> 5;
  const int hf = lane >> 4, ln = lane & 15;
  const int wm = wv >> 1, wn = wv & 1;
  const int mbase = blockIdx.y * 128, nbase = blockIdx.x * 128;

  // per-thread staging slots: 2 b128 for A, 2 for B per k-step
  const int srow0 = tid >> 2, skp0 = (tid & 3) * 8;
  const int srow1 = (tid + 256) >> 2, skp1 = ((tid + 256) & 3) * 8;

  v8f zf = {0.f, 0.f, 0.f, 0.f, 0.f, 0.f, 0.f, 0.f};
  v8f acc[2][4];
#pragma unroll
  for (int s = 0; s < 2; ++s)
#pragma unroll
    for (int b = 0; b < 4; ++b) acc[s][b] = zf;

  const int steps = K >> 5;

  // prologue: stage k-step 0 into buffer 0
  async_ld_b128((uint32_t)(uintptr_t)&Alds[0][srow0 * 32 + skp0],
                &A[(size_t)(mbase + srow0) * K + skp0]);
  async_ld_b128((uint32_t)(uintptr_t)&Alds[0][srow1 * 32 + skp1],
                &A[(size_t)(mbase + srow1) * K + skp1]);
  async_ld_b128((uint32_t)(uintptr_t)&Blds[0][srow0 * 32 + skp0],
                &Wt[(size_t)(nbase + srow0) * K + skp0]);
  async_ld_b128((uint32_t)(uintptr_t)&Blds[0][srow1 * 32 + skp1],
                &Wt[(size_t)(nbase + srow1) * K + skp1]);

  for (int t = 0; t < steps; ++t) {
    const int buf = t & 1;
    __syncthreads();  // all waves done reading the buffer we are about to overwrite
    if (t + 1 < steps) {
      const int nb = (t + 1) & 1, k0 = (t + 1) * 32;
      async_ld_b128((uint32_t)(uintptr_t)&Alds[nb][srow0 * 32 + skp0],
                    &A[(size_t)(mbase + srow0) * K + k0 + skp0]);
      async_ld_b128((uint32_t)(uintptr_t)&Alds[nb][srow1 * 32 + skp1],
                    &A[(size_t)(mbase + srow1) * K + k0 + skp1]);
      async_ld_b128((uint32_t)(uintptr_t)&Blds[nb][srow0 * 32 + skp0],
                    &Wt[(size_t)(nbase + srow0) * K + k0 + skp0]);
      async_ld_b128((uint32_t)(uintptr_t)&Blds[nb][srow1 * 32 + skp1],
                    &Wt[(size_t)(nbase + srow1) * K + k0 + skp1]);
      wait_async_le4();  // in-order completion => current buffer's 4 loads retired
    } else {
      wait_async_0();
    }
    __syncthreads();  // current buffer visible to all waves

    const uint32_t* A32 = (const uint32_t*)Alds[buf];
    const uint32_t* B32 = (const uint32_t*)Blds[buf];
    FragU afr[2], bfr[4];
#pragma unroll
    for (int sub = 0; sub < 2; ++sub) {
      int rowl = wm * 32 + sub * 16 + ln;
#pragma unroll
      for (int i = 0; i < 8; ++i) afr[sub].u[i] = A32[rowl * 16 + a_k2(i, hf)];
    }
#pragma unroll
    for (int bt = 0; bt < 4; ++bt) {
      int nl = wn * 64 + bt * 16 + ln;
#pragma unroll
      for (int i = 0; i < 8; ++i) bfr[bt].u[i] = B32[nl * 16 + a_k2(i, hf)];
    }
#pragma unroll
    for (int sub = 0; sub < 2; ++sub)
#pragma unroll
      for (int bt = 0; bt < 4; ++bt)
        acc[sub][bt] = wmma_bf16f32(afr[sub].v, bfr[bt].v, acc[sub][bt]);
  }

  // Epilogue: C layout — VGPR r: lanes 0-15 row r, lanes 16-31 row r+8; N = lane&15
#pragma unroll
  for (int sub = 0; sub < 2; ++sub)
#pragma unroll
    for (int bt = 0; bt < 4; ++bt)
#pragma unroll
      for (int r = 0; r < 8; ++r) {
        int row = mbase + wm * 32 + sub * 16 + hf * 8 + r;
        int col = nbase + wn * 64 + bt * 16 + ln;
        float v = acc[sub][bt][r] + bias[col];
        if (relu) v = fmaxf(v, 0.f);
        if (outF) outF[(size_t)row * N + col] = v;
        if (outB) outB[(size_t)row * N + col] = (__bf16)v;
      }
}

// ---------------------------------------------------------------- attention
// Flash-style: 8 waves/WG, wave = one 16-row Q tile; stream 32-key K/V chunks.
__global__ __launch_bounds__(256) void attn_kernel(
    const __bf16* __restrict__ Q, const __bf16* __restrict__ Kmat,
    const __bf16* __restrict__ Vmat, const int* __restrict__ mask,
    __bf16* __restrict__ Ob, int S) {
  __shared__ __bf16   Klds[32 * 64];   // [key][dk]
  __shared__ uint32_t Vsw[16 * 64];    // [key/2][dk] pair-packed {V[2r][d],V[2r+1][d]}
  __shared__ __bf16   Plds[8][16 * 32];

  const int tid = threadIdx.x;
  const int lane = tid & 31, wv = tid >> 5;
  const int hf = lane >> 4, ln = lane & 15;
  const int b = blockIdx.z, h = blockIdx.y;
  const int q0 = blockIdx.x * 128 + wv * 16;
  const size_t base = (size_t)b * S * D_MODEL + h * HEAD_DK;

  // Preload Q fragments (two K-chunks of 32 over DK=64)
  FragU qf[2];
#pragma unroll
  for (int kc = 0; kc < 2; ++kc)
#pragma unroll
    for (int i = 0; i < 8; ++i) {
      int k2 = a_k2(i, hf) + kc * 16;
      qf[kc].u[i] = *(const uint32_t*)&Q[base + (size_t)(q0 + ln) * D_MODEL + k2 * 2];
    }

  v8f zf = {0.f, 0.f, 0.f, 0.f, 0.f, 0.f, 0.f, 0.f};
  v8f o[4];
  o[0] = zf; o[1] = zf; o[2] = zf; o[3] = zf;
  float m8[8], l8[8];
#pragma unroll
  for (int r = 0; r < 8; ++r) { m8[r] = -3.0e38f; l8[r] = 0.f; }

  for (int kb = 0; kb < S; kb += 32) {
    __syncthreads();
    {  // K chunk [32][64] via async global->LDS
      int row = tid >> 3, dp = (tid & 7) * 8;
      async_ld_b128((uint32_t)(uintptr_t)&Klds[row * 64 + dp],
                    &Kmat[base + (size_t)(kb + row) * D_MODEL + dp]);
    }
    {  // V chunk pair-packed (needs a swizzle, so manual)
      int r = tid >> 4, dp = (tid & 15) * 4;
      const uint32_t* v0 = (const uint32_t*)&Vmat[base + (size_t)(kb + 2 * r) * D_MODEL + dp];
      const uint32_t* v1 = (const uint32_t*)&Vmat[base + (size_t)(kb + 2 * r + 1) * D_MODEL + dp];
      uint32_t a0 = v0[0], a1 = v1[0], b0 = v0[1], b1 = v1[1];
      uint4 q = {(a0 & 0xFFFFu) | (a1 << 16), (a0 >> 16) | (a1 & 0xFFFF0000u),
                 (b0 & 0xFFFFu) | (b1 << 16), (b0 >> 16) | (b1 & 0xFFFF0000u)};
      *(uint4*)&Vsw[r * 64 + dp] = q;
    }
    wait_async_0();
    __syncthreads();

    // S = Q * K^T over DK=64 for two 16-key tiles
    v8f s[2];
    s[0] = zf; s[1] = zf;
    const uint32_t* K32 = (const uint32_t*)Klds;
#pragma unroll
    for (int kt = 0; kt < 2; ++kt) {
#pragma unroll
      for (int kc = 0; kc < 2; ++kc) {
        FragU kf;
#pragma unroll
        for (int i = 0; i < 8; ++i)
          kf.u[i] = K32[(kt * 16 + ln) * 32 + kc * 16 + a_k2(i, hf)];
        s[kt] = wmma_bf16f32(qf[kc].v, kf.v, s[kt]);
      }
      int mk = mask[(size_t)b * S + kb + kt * 16 + ln];
      float madd = mk ? 0.f : -1e9f;
#pragma unroll
      for (int r = 0; r < 8; ++r) s[kt][r] = s[kt][r] * 0.125f + madd;
    }

    // Online softmax (row groups of 16 lanes)
#pragma unroll
    for (int r = 0; r < 8; ++r) {
      float t = fmaxf(s[0][r], s[1][r]);
#pragma unroll
      for (int mset = 8; mset >= 1; mset >>= 1) t = fmaxf(t, __shfl_xor(t, mset, 32));
      float mnew = fmaxf(m8[r], t);
      float alpha = __expf(m8[r] - mnew);
      float p0 = __expf(s[0][r] - mnew);
      float p1 = __expf(s[1][r] - mnew);
      float rs = p0 + p1;
#pragma unroll
      for (int mset = 8; mset >= 1; mset >>= 1) rs += __shfl_xor(rs, mset, 32);
      l8[r] = l8[r] * alpha + rs;
      m8[r] = mnew;
      Plds[wv][(hf * 8 + r) * 32 + ln]      = (__bf16)p0;
      Plds[wv][(hf * 8 + r) * 32 + 16 + ln] = (__bf16)p1;
#pragma unroll
      for (int nt = 0; nt < 4; ++nt) o[nt][r] *= alpha;
    }

    // O += P * V  (P re-read as A-fragment; wave-local LDS, DS in-order)
    FragU pf;
    const uint32_t* P32 = (const uint32_t*)Plds[wv];
#pragma unroll
    for (int i = 0; i < 8; ++i) pf.u[i] = P32[ln * 16 + a_k2(i, hf)];
#pragma unroll
    for (int nt = 0; nt < 4; ++nt) {
      FragU vf;
#pragma unroll
      for (int i = 0; i < 8; ++i) vf.u[i] = Vsw[a_k2(i, hf) * 64 + nt * 16 + ln];
      o[nt] = wmma_bf16f32(pf.v, vf.v, o[nt]);
    }
  }

#pragma unroll
  for (int r = 0; r < 8; ++r) {
    float inv = l8[r] > 0.f ? 1.f / l8[r] : 0.f;
    size_t row = (size_t)b * S + q0 + hf * 8 + r;
#pragma unroll
    for (int nt = 0; nt < 4; ++nt)
      Ob[row * D_MODEL + h * HEAD_DK + nt * 16 + ln] = (__bf16)(o[nt][r] * inv);
  }
}

// ---------------------------------------------------------------- residual+LN
__global__ __launch_bounds__(256) void add_ln_kernel(
    const float* __restrict__ X, const float* __restrict__ R,
    const float* __restrict__ g, const float* __restrict__ be,
    float* __restrict__ outF, __bf16* __restrict__ outB) {
  __shared__ float sm[8];
  const int tid = threadIdx.x;
  const size_t row = blockIdx.x;
  const int c0 = tid * 4;

  float4 xv = *(const float4*)(X + row * D_MODEL + c0);
  float4 rv = *(const float4*)(R + row * D_MODEL + c0);
  float v[4] = {xv.x + rv.x, xv.y + rv.y, xv.z + rv.z, xv.w + rv.w};

  float s = v[0] + v[1] + v[2] + v[3];
#pragma unroll
  for (int m = 16; m >= 1; m >>= 1) s += __shfl_xor(s, m, 32);
  if ((tid & 31) == 0) sm[tid >> 5] = s;
  __syncthreads();
  float tot = 0.f;
#pragma unroll
  for (int j = 0; j < 8; ++j) tot += sm[j];
  float mean = tot * (1.f / D_MODEL);
  __syncthreads();

  float vs = 0.f;
#pragma unroll
  for (int j = 0; j < 4; ++j) { float d = v[j] - mean; vs += d * d; }
#pragma unroll
  for (int m = 16; m >= 1; m >>= 1) vs += __shfl_xor(vs, m, 32);
  if ((tid & 31) == 0) sm[tid >> 5] = vs;
  __syncthreads();
  tot = 0.f;
#pragma unroll
  for (int j = 0; j < 8; ++j) tot += sm[j];
  float rstd = rsqrtf(tot * (1.f / D_MODEL) + EPS_LN);

#pragma unroll
  for (int j = 0; j < 4; ++j) {
    float ov = (v[j] - mean) * rstd * g[c0 + j] + be[c0 + j];
    if (outF) outF[row * D_MODEL + c0 + j] = ov;
    if (outB) outB[row * D_MODEL + c0 + j] = (__bf16)ov;
  }
}

// ---------------------------------------------------------------- host side
static inline void* ws_take(char*& p, size_t bytes) {
  void* r = (void*)p;
  p += (bytes + 255) & ~(size_t)255;
  return r;
}

static void launch_gemm(const __bf16* A, const __bf16* Wt, const float* bias,
                        float* outF, __bf16* outB, int M, int N, int K, int relu,
                        hipStream_t s) {
  dim3 grid(N / 128, M / 128);
  gemm_bf16_kernel<<<grid, 256, 0, s>>>(A, Wt, bias, outF, outB, M, N, K, relu);
}

static void launch_f2bf(const float* src, __bf16* dst, int n, hipStream_t s) {
  f2bf_kernel<<<(n / 4 + 255) / 256, 256, 0, s>>>(src, dst, n);
}

static void launch_f2bf_t(const float* src, __bf16* dst, int K, int N, hipStream_t s) {
  dim3 grid(N / 32, K / 32);
  f2bf_t_kernel<<<grid, 256, 0, s>>>(src, dst, K, N);
}

extern "C" void kernel_launch(void* const* d_in, const int* in_sizes, int n_in,
                              void* d_out, int out_size, void* d_ws, size_t ws_size,
                              hipStream_t stream) {
  const float* x    = (const float*)d_in[0];
  const float* y    = (const float*)d_in[1];
  const int*   mask = (const int*)d_in[2];
  const float* Wq = (const float*)d_in[3];
  const float* bq = (const float*)d_in[4];
  const float* Wk = (const float*)d_in[5];
  const float* bk = (const float*)d_in[6];
  const float* Wv = (const float*)d_in[7];
  const float* bv = (const float*)d_in[8];
  const float* Wo = (const float*)d_in[9];
  const float* bo = (const float*)d_in[10];
  const float* W1 = (const float*)d_in[11];
  const float* b1 = (const float*)d_in[12];
  const float* W2 = (const float*)d_in[13];
  const float* b2 = (const float*)d_in[14];
  const float* g1 = (const float*)d_in[15];
  const float* be1 = (const float*)d_in[16];
  const float* g2 = (const float*)d_in[17];
  const float* be2 = (const float*)d_in[18];

  const int BS = in_sizes[0] / D_MODEL;  // B*S = 4096
  const int S = SEQ_LEN;
  const int B = BS / S;

  char* wp = (char*)d_ws;
  __bf16* xb  = (__bf16*)ws_take(wp, (size_t)BS * D_MODEL * 2);
  __bf16* yb  = (__bf16*)ws_take(wp, (size_t)BS * D_MODEL * 2);
  __bf16* Wqt = (__bf16*)ws_take(wp, (size_t)D_MODEL * D_MODEL * 2);
  __bf16* Wkt = (__bf16*)ws_take(wp, (size_t)D_MODEL * D_MODEL * 2);
  __bf16* Wvt = (__bf16*)ws_take(wp, (size_t)D_MODEL * D_MODEL * 2);
  __bf16* Wot = (__bf16*)ws_take(wp, (size_t)D_MODEL * D_MODEL * 2);
  __bf16* W1t = (__bf16*)ws_take(wp, (size_t)D_MODEL * HID_DIM * 2);
  __bf16* W2t = (__bf16*)ws_take(wp, (size_t)HID_DIM * D_MODEL * 2);
  // Q/K/V/attn region (32 MB) is later reused for the FFN hidden activation.
  __bf16* Qb    = (__bf16*)ws_take(wp, (size_t)BS * D_MODEL * 2);
  __bf16* Kb    = (__bf16*)ws_take(wp, (size_t)BS * D_MODEL * 2);
  __bf16* Vb    = (__bf16*)ws_take(wp, (size_t)BS * D_MODEL * 2);
  __bf16* attnb = (__bf16*)ws_take(wp, (size_t)BS * D_MODEL * 2);
  float*  tmpF  = (float*)ws_take(wp, (size_t)BS * D_MODEL * 4);  // mha_out, then ffn_out
  float*  x1f   = (float*)ws_take(wp, (size_t)BS * D_MODEL * 4);
  __bf16* x1b   = xb;        // xb dead after QKV projections
  __bf16* h1b   = Qb;        // Q/K/V/attn dead after O-projection (32 MB span)

  // 1) conversions: activations plain, weights transposed to [N,K] bf16
  launch_f2bf(x, xb, BS * D_MODEL, stream);
  launch_f2bf(y, yb, BS * D_MODEL, stream);
  launch_f2bf_t(Wq, Wqt, D_MODEL, D_MODEL, stream);
  launch_f2bf_t(Wk, Wkt, D_MODEL, D_MODEL, stream);
  launch_f2bf_t(Wv, Wvt, D_MODEL, D_MODEL, stream);
  launch_f2bf_t(Wo, Wot, D_MODEL, D_MODEL, stream);
  launch_f2bf_t(W1, W1t, D_MODEL, HID_DIM, stream);
  launch_f2bf_t(W2, W2t, HID_DIM, D_MODEL, stream);

  // 2) QKV projections
  launch_gemm(xb, Wqt, bq, nullptr, Qb, BS, D_MODEL, D_MODEL, 0, stream);
  launch_gemm(yb, Wkt, bk, nullptr, Kb, BS, D_MODEL, D_MODEL, 0, stream);
  launch_gemm(yb, Wvt, bv, nullptr, Vb, BS, D_MODEL, D_MODEL, 0, stream);

  // 3) flash attention (output already merged back to [B*S, D])
  dim3 agrid(S / 128, N_HEADS, B);
  attn_kernel<<<agrid, 256, 0, stream>>>(Qb, Kb, Vb, mask, attnb, S);

  // 4) output projection, 5) residual + LN1
  launch_gemm(attnb, Wot, bo, tmpF, nullptr, BS, D_MODEL, D_MODEL, 0, stream);
  add_ln_kernel<<<BS, 256, 0, stream>>>(x, tmpF, g1, be1, x1f, x1b);

  // 6) FFN1 (+ReLU), 7) FFN2, 8) residual + LN2 -> d_out
  launch_gemm(x1b, W1t, b1, nullptr, h1b, BS, HID_DIM, D_MODEL, 1, stream);
  launch_gemm(h1b, W2t, b2, tmpF, nullptr, BS, D_MODEL, HID_DIM, 0, stream);
  add_ln_kernel<<<BS, 256, 0, stream>>>(x1f, tmpF, g2, be2, (float*)d_out, nullptr);

  (void)n_in; (void)out_size; (void)ws_size;
}
